// Offset_Gen_34291018891858
// MI455X (gfx1250) — matrix operand order repaired and checked by reading, not compiled, not run on hardware
//
#include <hip/hip_runtime.h>
#include <hip/hip_bf16.h>
#include <stdint.h>

// MI455X / gfx1250 fused kernel for windowed-QK attention-score generation.
// Wave32, WMMA f32<-f16 16x16x32, fp32 accumulation throughout.

typedef __attribute__((ext_vector_type(16))) _Float16 v16h;
typedef __attribute__((ext_vector_type(8)))  float    v8f;

#define DIM    64
#define QK_OUT 320
#define NQ     256   // N * HEADS (q channels)
#define HEADS  4
#define CK     16
#define XSTR   72    // padded f16 row stride (64+8): 144B rows, 16B-aligned, bank-spread
#define WSTR   72

// ---- A-operand fragment (16x32 f16, MxK). ISA 7.12.2:
//   lanes 0-15 / 16-31 both hold M = lane&15;
//   VGPR v holds f16 pair K = k0 + (v&4?16:0) + (lane>=16?8:0) + 2*(v&3).
// base layout: row-major [m][k], K contiguous -> one b32 LDS load per VGPR.
__device__ __forceinline__ v16h lds_load_A(const _Float16* base, int row0,
                                           int stride, int k0, int lane) {
  const int m  = lane & 15;
  const int hi = (lane >> 4) & 1;
  v16h a;
#pragma unroll
  for (int v = 0; v < 8; ++v) {
    const int k = k0 + ((v & 4) ? 16 : 0) + (hi ? 8 : 0) + 2 * (v & 3);
    union { uint32_t u; _Float16 h[2]; } t;
    t.u = *(const uint32_t*)(base + (size_t)(row0 + m) * stride + k);
    a[2 * v]     = t.h[0];
    a[2 * v + 1] = t.h[1];
  }
  return a;
}

// ---- B-operand fragment (32x16 f16, KxN). N = lane&15 across lanes;
//   VGPR v holds f16 pair K = k0 + (v&4?16:0) + (lane>=16?8:0) + 2*(v&3) (matches A).
// base layout: [n][k] (K contiguous per output column) -> one b32 load per VGPR.
__device__ __forceinline__ v16h lds_load_B(const _Float16* base, int col0,
                                           int stride, int k0, int lane) {
  const int n  = lane & 15;
  const int hi = (lane >> 4) & 1;
  v16h b;
#pragma unroll
  for (int v = 0; v < 8; ++v) {
    const int k = k0 + ((v & 4) ? 16 : 0) + (hi ? 8 : 0) + 2 * (v & 3);
    union { uint32_t u; _Float16 h[2]; } t;
    t.u = *(const uint32_t*)(base + (size_t)(col0 + n) * stride + k);
    b[2 * v]     = t.h[0];
    b[2 * v + 1] = t.h[1];
  }
  return b;
}

// One-time fp32 -> f16 conversion of W_qk (320x64) into workspace.
__global__ void wqk_to_f16_kernel(const float* __restrict__ W,
                                  _Float16* __restrict__ Wh, int n) {
  const int i = blockIdx.x * blockDim.x + threadIdx.x;
  if (i < n) Wh[i] = (_Float16)W[i];
}

// One block per (b, h1, w1) group: 8*32*32 = 8192 blocks, 256 threads (8 waves).
__global__ __launch_bounds__(256)
void fused_window_qk_attn_kernel(const float* __restrict__ x,
                                 const _Float16* __restrict__ Wh,
                                 const float* __restrict__ bias,
                                 const float* __restrict__ temp,
                                 float* __restrict__ out) {
  extern __shared__ _Float16 smem[];
  _Float16* Xs  = smem;               // 64  x XSTR f16 (gathered x tile, [p][c])
  _Float16* Ws  = smem + 64 * XSTR;   // 320 x WSTR f16 (weights, [o][c])
  _Float16* QKs = Ws;                 // aliased after GEMM1: QKt [o][p]

  const int blk = blockIdx.x;
  const int b   = blk >> 10;          // / (32*32)
  const int h1  = (blk >> 5) & 31;
  const int w1  = blk & 31;

  const int tid  = threadIdx.x;
  const int lane = tid & 31;
  const int wave = tid >> 5;

  // ---- Phase 0a: gather 64 dilated positions (ph*32+h1, pw*32+w1) of x,
  //      convert fp32 -> f16 into LDS X[p][c]. 4 threads per position.
  {
    const int pos = tid >> 2;         // 0..63, p = ph*8 + pw
    const int q4  = tid & 3;
    const int ph = pos >> 3, pw = pos & 7;
    const int gy = ph * 32 + h1, gx = pw * 32 + w1;
    const float4* xrow =
        (const float4*)(x + (((size_t)b * 256 + gy) * 256 + gx) * DIM);
#pragma unroll
    for (int i = 0; i < 4; ++i) {
      const int f = q4 * 4 + i;       // float4 index within the 64-ch row
      const float4 val = xrow[f];
      union { _Float16 h[4]; uint2 u2; } t;
      t.h[0] = (_Float16)val.x; t.h[1] = (_Float16)val.y;
      t.h[2] = (_Float16)val.z; t.h[3] = (_Float16)val.w;
      *(uint2*)(Xs + (size_t)pos * XSTR + f * 4) = t.u2;
    }
  }

  // ---- Phase 0b: stage f16 W (320x64, L2-resident) into padded LDS rows.
  {
    const uint4* wsrc = (const uint4*)Wh;   // 2560 16-byte chunks
#pragma unroll
    for (int j = 0; j < 10; ++j) {
      const int u  = tid + 256 * j;
      const uint4 v4 = wsrc[u];
      const int o = u >> 3, c8 = u & 7;
      *(uint4*)((char*)Ws + (size_t)o * (WSTR * 2) + c8 * 16) = v4;
    }
  }
  __syncthreads();

  // ---- GEMM1: QK[p][o] = X[p][:] . W[o][:]   (M=64, N=320, K=64)
  // 4x20 tile grid; wave -> mtile = wave/2, 10 ntiles each.
  const int mtile = wave >> 1;
  const int ntb   = (wave & 1) * 10;
  const v16h a0 = lds_load_A(Xs, mtile * 16, XSTR, 0,  lane);
  const v16h a1 = lds_load_A(Xs, mtile * 16, XSTR, 32, lane);
  v8f acc[10];
#pragma unroll
  for (int t = 0; t < 10; ++t) {
    const int o0 = (ntb + t) * 16;
    const v16h b0 = lds_load_B(Ws, o0, WSTR, 0,  lane);
    const v16h b1 = lds_load_B(Ws, o0, WSTR, 32, lane);
    v8f c = {};
    c = __builtin_amdgcn_wmma_f32_16x16x32_f16(false, a0, false, b0,
                                               (short)0, c, false, false);
    c = __builtin_amdgcn_wmma_f32_16x16x32_f16(false, a1, false, b1,
                                               (short)0, c, false, false);
    acc[t] = c;
  }
  __syncthreads();   // all Ws reads retired -> safe to alias QKs over Ws

  // ---- Bias add + pack to f16, store transposed QKt[o][p].
  // D layout: lane holds n=lane&15 fixed, m(=p) = v + 8*(lane>=16):
  // consecutive VGPRs v,v+1 are consecutive p -> b32 pair stores.
  {
    const int hi = (lane >> 4) & 1;
    const int oc = lane & 15;
#pragma unroll
    for (int t = 0; t < 10; ++t) {
      const int o  = (ntb + t) * 16 + oc;
      const float bv = bias[o];
#pragma unroll
      for (int v = 0; v < 8; v += 2) {
        union { uint32_t u; _Float16 h[2]; } pk;
        pk.h[0] = (_Float16)(acc[t][v]     + bv);
        pk.h[1] = (_Float16)(acc[t][v + 1] + bv);
        const int p = mtile * 16 + hi * 8 + v;
        *(uint32_t*)(QKs + (size_t)o * WSTR + p) = pk.u;
      }
    }
  }
  __syncthreads();

  // ---- GEMM2 per head: attn[n,d] = sum_p QKt[head*64+n][p] * QKt[256+head*16+d][p]
  // 16 tiles (4 heads x 4 n-tiles x 1 d-tile), 2 per wave. Scale + scatter out.
#pragma unroll
  for (int tt = 0; tt < 2; ++tt) {
    const int t    = wave * 2 + tt;
    const int head = t >> 2;
    const int mt   = t & 3;
    const v16h A0 = lds_load_A(QKs, head * 64 + mt * 16, WSTR, 0,  lane);
    const v16h A1 = lds_load_A(QKs, head * 64 + mt * 16, WSTR, 32, lane);
    const v16h B0 = lds_load_B(QKs, NQ + head * CK, WSTR, 0,  lane);
    const v16h B1 = lds_load_B(QKs, NQ + head * CK, WSTR, 32, lane);
    v8f c = {};
    c = __builtin_amdgcn_wmma_f32_16x16x32_f16(false, A0, false, B0,
                                               (short)0, c, false, false);
    c = __builtin_amdgcn_wmma_f32_16x16x32_f16(false, A1, false, B1,
                                               (short)0, c, false, false);
    const float ts = temp[head];
    const int d  = lane & 15;
    const int hi = (lane >> 4) & 1;
    const int ch = head * CK + d;
#pragma unroll
    for (int v = 0; v < 8; ++v) {
      const int n  = mt * 16 + hi * 8 + v;      // n = ws_a*8 + ws_b
      const int ya = h1 * 8 + (n >> 3);
      const int xb = w1 * 8 + (n & 7);
      out[(((size_t)b * 256 + ya) * 256 + xb) * 64 + ch] = c[v] * ts;
    }
  }
}

extern "C" void kernel_launch(void* const* d_in, const int* in_sizes, int n_in,
                              void* d_out, int out_size, void* d_ws, size_t ws_size,
                              hipStream_t stream) {
  const float* x    = (const float*)d_in[0];   // (8,256,256,64)
  const float* Wqk  = (const float*)d_in[1];   // (320,64)
  const float* bqk  = (const float*)d_in[2];   // (320,)
  const float* temp = (const float*)d_in[3];   // (4,)
  float* out = (float*)d_out;                  // (8,256,256,64)

  _Float16* Wh = (_Float16*)d_ws;              // 320*64 f16 = 40 KB scratch

  // One-time (per launch, deterministic) weight conversion to f16.
  wqk_to_f16_kernel<<<(QK_OUT * DIM + 255) / 256, 256, 0, stream>>>(
      Wqk, Wh, QK_OUT * DIM);

  // 8 batches * 32 * 32 groups; 55296 B dynamic LDS (X tile + W/QK union).
  const int    grid = 8 * 32 * 32;
  const size_t lds  = (size_t)(64 * XSTR + QK_OUT * WSTR) * sizeof(_Float16);
  fused_window_qk_attn_kernel<<<grid, 256, lds, stream>>>(x, Wh, bqk, temp, out);
}
